// LFM2_4818953306187
// MI455X (gfx1250) — matrix operand order repaired and checked by reading, not compiled, not run on hardware
//
#include <hip/hip_runtime.h>
#include <hip/hip_bf16.h>

#define Hd   768
#define NHd  12
#define NKVd 4
#define HDd  64
#define Vd   50304
#define IMd  3072
#define Bd   2
#define Sd   1024
#define Td   (Bd*Sd)   // 2048 tokens

typedef _Float16 h16v __attribute__((ext_vector_type(16)));
typedef _Float16 h8v  __attribute__((ext_vector_type(8)));
typedef _Float16 h4v  __attribute__((ext_vector_type(4)));
typedef float    f8v  __attribute__((ext_vector_type(8)));
typedef float    f4v  __attribute__((ext_vector_type(4)));
typedef unsigned int u32x4 __attribute__((ext_vector_type(4)));
typedef int          i32x4 __attribute__((ext_vector_type(4)));
typedef int          i32x8 __attribute__((ext_vector_type(8)));

#if defined(__gfx1250__) && __has_builtin(__builtin_amdgcn_tensor_load_to_lds) && \
    __has_builtin(__builtin_amdgcn_s_wait_tensorcnt)
#define USE_TDM 1
#else
#define USE_TDM 0
#endif

static __device__ __forceinline__ f8v wmma16(h16v a, h16v b, f8v c) {
  return __builtin_amdgcn_wmma_f32_16x16x32_f16(false, a, false, b, (short)0, c,
                                                false, false);
}
static __device__ __forceinline__ h16v cat8(h8v lo, h8v hi) {
  return __builtin_shufflevector(lo, hi, 0,1,2,3,4,5,6,7,8,9,10,11,12,13,14,15);
}
static __device__ __forceinline__ unsigned ldsoff(const void* p) {
  return (unsigned)(unsigned long long)p;   // low 32 bits of LDS generic addr
}

#if USE_TDM
// 2D f16 tile DMA: global (row_stride elems between rows) -> contiguous LDS.
// D# per CDNA5 ISA 8.3/8.4: group0 {count=1, lds_addr, global_addr, type=2},
// group1 {data_size=2B, tensor dims, tile_dim0/1, dim0 stride}.
static __device__ __forceinline__ void tdm_load_tile_f16(
    const _Float16* gsrc, unsigned lds_byte_off, int tile_w, int tile_h,
    long long row_stride_elems) {
  unsigned long long ga = (unsigned long long)(const void*)gsrc;
  u32x4 g0;
  g0.x = 1u;                                                 // count = 1
  g0.y = lds_byte_off;                                       // lds_addr
  g0.z = (unsigned)ga;                                       // global_addr lo
  g0.w = (unsigned)((ga >> 32) & 0x01ffffffu) | (2u << 30);  // addr hi, type=2
  unsigned td0 = (unsigned)row_stride_elems;                 // tensor_dim0
  unsigned td1 = 0x00ffffffu;                                // tensor_dim1 (ample)
  unsigned long long st0 = (unsigned long long)row_stride_elems;
  i32x8 g1;
  g1[0] = 1 << 16;                                           // data_size = 2B
  g1[1] = (int)((td0 & 0xffffu) << 16);
  g1[2] = (int)(((td0 >> 16) & 0xffffu) | ((td1 & 0xffffu) << 16));
  g1[3] = (int)(((td1 >> 16) & 0xffffu) | (((unsigned)tile_w & 0xffffu) << 16));
  g1[4] = (int)((unsigned)tile_h & 0xffffu);                 // tile_dim1, dim2=0
  g1[5] = (int)(unsigned)(st0 & 0xffffffffu);                // dim0 stride lo
  g1[6] = (int)(unsigned)((st0 >> 32) & 0xffffu);            // stride hi
  g1[7] = 0;
  i32x4 z4 = {0, 0, 0, 0};
#if __clang_major__ >= 23
  i32x8 z8 = {0, 0, 0, 0, 0, 0, 0, 0};
  __builtin_amdgcn_tensor_load_to_lds(g0, g1, z4, z4, z8, 0);
#else
  __builtin_amdgcn_tensor_load_to_lds(g0, g1, z4, z4, 0);
#endif
}
#endif

// ---------------------------------------------------------------------------
// f32 -> f16 weight conversion
// ---------------------------------------------------------------------------
__global__ void lfm2_cvt_f16(const float* __restrict__ src,
                             _Float16* __restrict__ dst, size_t n) {
  size_t i = (size_t)blockIdx.x * 256 + threadIdx.x;
  if (i < n) dst[i] = (_Float16)src[i];
}

// ---------------------------------------------------------------------------
// Tiled WMMA GEMM:  C[M,N] = A[M,K](f32) @ W[N,K]^T(f16)  [+bias] [+resid]
// 128x128x64 tiles, 256 threads (8 waves), each wave 32x64 (2x4 WMMA frags).
// Double-buffered LDS; W tile via TDM (tensor_load_to_lds), A tile staged
// through registers with on-the-fly f32->f16 conversion.
// ---------------------------------------------------------------------------
#define GBM 128
#define GBN 128
#define GBK 64

__global__ __launch_bounds__(256)
void lfm2_gemm(const float* __restrict__ A, const _Float16* __restrict__ W,
               float* C, const float* __restrict__ bias, const float* resid,
               int M, int N, int K) {
  __shared__ _Float16 As[2][GBM][GBK];
  __shared__ _Float16 Bs[2][GBN][GBK];
  const int tid  = threadIdx.x;
  const int wid  = tid >> 5;
  const int lane = tid & 31;
  const int g    = (lane >> 4) & 1;
  const int lr   = lane & 15;
  const int wm   = wid & 3;       // 4 row-waves of 32
  const int wn   = wid >> 2;      // 2 col-waves of 64
  const int bm0  = blockIdx.y * GBM;
  const int bn0  = blockIdx.x * GBN;
  const int ar   = tid >> 1;          // staging row 0..127
  const int ac   = (tid & 1) << 5;    // staging col 0 or 32

  f8v acc[2][4];
  for (int i = 0; i < 2; ++i)
    for (int j = 0; j < 4; ++j) acc[i][j] = (f8v){0,0,0,0,0,0,0,0};

  const int nk = K / GBK;
  f4v areg[8];
#if !USE_TDM
  h8v breg[4];
#endif
  // ---- prologue: stage tile 0 ----
  {
    const float* ap = A + (size_t)(bm0 + ar) * K + ac;
    for (int p = 0; p < 8; ++p) areg[p] = *(const f4v*)(ap + p * 4);
#if USE_TDM
    tdm_load_tile_f16(W + (size_t)(bn0 + wid * 16) * K,
                      ldsoff(&Bs[0][wid * 16][0]), GBK, 16, K);
#else
    const _Float16* wp = W + (size_t)(bn0 + ar) * K + ac;
    for (int p = 0; p < 4; ++p) breg[p] = *(const h8v*)(wp + p * 8);
#endif
  }
  for (int p = 0; p < 8; ++p) {
    f4v v = areg[p];
    h4v hv; hv.x = (_Float16)v.x; hv.y = (_Float16)v.y;
            hv.z = (_Float16)v.z; hv.w = (_Float16)v.w;
    *(h4v*)&As[0][ar][ac + p * 4] = hv;
  }
#if USE_TDM
  __builtin_amdgcn_s_wait_tensorcnt(0);
#else
  for (int p = 0; p < 4; ++p) *(h8v*)&Bs[0][ar][ac + p * 8] = breg[p];
#endif
  __syncthreads();

  for (int kt = 0; kt < nk; ++kt) {
    const int cur = kt & 1, nxt = cur ^ 1;
    const bool more = (kt + 1) < nk;
    if (more) {       // issue next tile's traffic before computing this one
      const float* ap = A + (size_t)(bm0 + ar) * K + (kt + 1) * GBK + ac;
      for (int p = 0; p < 8; ++p) areg[p] = *(const f4v*)(ap + p * 4);
#if USE_TDM
      tdm_load_tile_f16(W + (size_t)(bn0 + wid * 16) * K + (kt + 1) * GBK,
                        ldsoff(&Bs[nxt][wid * 16][0]), GBK, 16, K);
#else
      const _Float16* wp = W + (size_t)(bn0 + ar) * K + (kt + 1) * GBK + ac;
      for (int p = 0; p < 4; ++p) breg[p] = *(const h8v*)(wp + p * 8);
#endif
      if (kt + 2 < nk) {  // warm L2 for the tile after next
        __builtin_prefetch(A + (size_t)(bm0 + ar) * K + (kt + 2) * GBK + ac, 0, 1);
        __builtin_prefetch(W + (size_t)(bn0 + ar) * K + (kt + 2) * GBK + ac, 0, 1);
      }
    }
    // ---- compute on LDS[cur] ----
    for (int kk = 0; kk < 2; ++kk) {
      h16v af[2], bf[4];
      for (int i = 0; i < 2; ++i) {   // A frag: a[j<8]=A[m][8g+j], a[j>=8]=A[m][16+8g+j-8]
        int m = wm * 32 + i * 16 + lr;
        h8v lo = *(const h8v*)&As[cur][m][kk * 32 + 8 * g];
        h8v hi = *(const h8v*)&As[cur][m][kk * 32 + 16 + 8 * g];
        af[i] = cat8(lo, hi);
      }
      for (int j = 0; j < 4; ++j) {   // B frag: b[j<8]=W[n][8g+j], b[j>=8]=W[n][16+8g+j-8]
        int n = wn * 64 + j * 16 + lr;
        h8v lo = *(const h8v*)&Bs[cur][n][kk * 32 + 8 * g];
        h8v hi = *(const h8v*)&Bs[cur][n][kk * 32 + 16 + 8 * g];
        bf[j] = cat8(lo, hi);
      }
      for (int i = 0; i < 2; ++i)
        for (int j = 0; j < 4; ++j) acc[i][j] = wmma16(af[i], bf[j], acc[i][j]);
    }
    if (more) {       // commit next tile
      for (int p = 0; p < 8; ++p) {
        f4v v = areg[p];
        h4v hv; hv.x = (_Float16)v.x; hv.y = (_Float16)v.y;
                hv.z = (_Float16)v.z; hv.w = (_Float16)v.w;
        *(h4v*)&As[nxt][ar][ac + p * 4] = hv;
      }
#if USE_TDM
      __builtin_amdgcn_s_wait_tensorcnt(0);
#else
      for (int p = 0; p < 4; ++p) *(h8v*)&Bs[nxt][ar][ac + p * 8] = breg[p];
#endif
    }
    __syncthreads();
  }
  // ---- epilogue: C layout M = r + 8*(lane/16), N = lane%16 ----
  for (int i = 0; i < 2; ++i)
    for (int j = 0; j < 4; ++j) {
      int gn = bn0 + wn * 64 + j * 16 + lr;
      float bv = bias ? bias[gn] : 0.0f;
      for (int r = 0; r < 8; ++r) {
        int gm = bm0 + wm * 32 + i * 16 + 8 * g + r;
        size_t o = (size_t)gm * N + gn;
        float v = acc[i][j][r] + bv;
        if (resid) v += resid[o];
        C[o] = v;
      }
    }
}

// ---------------------------------------------------------------------------
// Flash attention, 4 waves per (b, h, 64-query block), WMMA QK^T and PV
// Qh:[B,NH,S,HD] f16, Kh/Vh:[B,NKV,S,HD] f16, O:[B,S,H] f32
// ---------------------------------------------------------------------------
__global__ __launch_bounds__(128)
void lfm2_attn(const _Float16* __restrict__ Qh, const _Float16* __restrict__ Kh,
               const _Float16* __restrict__ Vh, float* __restrict__ O) {
  __shared__ _Float16 Qs[64][64];
  __shared__ _Float16 Kt[64][64];
  __shared__ _Float16 Vt[64][64];      // transposed: [dim][key]
  __shared__ _Float16 Pw[4][16][64];
  const int qb = blockIdx.x, hh = blockIdx.y, b = blockIdx.z;
  const int kvh = hh / (NHd / NKVd);
  const int tid = threadIdx.x, w = tid >> 5, lane = tid & 31;
  const int g = (lane >> 4) & 1, lr = lane & 15;
  const _Float16* Qb = Qh + (((size_t)(b * NHd  + hh )) * Sd + qb * 64) * HDd;
  const _Float16* Kb = Kh + (((size_t)(b * NKVd + kvh)) * Sd) * HDd;
  const _Float16* Vb = Vh + (((size_t)(b * NKVd + kvh)) * Sd) * HDd;

  for (int p = 0; p < 4; ++p) {
    int idx = p * 128 + tid;
    int r = idx >> 3, c = (idx & 7) << 3;
    *(h8v*)&Qs[r][c] = *(const h8v*)(Qb + r * HDd + c);
  }
  __syncthreads();
  h16v qa[2];
  for (int kk = 0; kk < 2; ++kk) {
    h8v lo = *(const h8v*)&Qs[w * 16 + lr][kk * 32 + 8 * g];
    h8v hi = *(const h8v*)&Qs[w * 16 + lr][kk * 32 + 16 + 8 * g];
    qa[kk] = cat8(lo, hi);
  }
  float mrun[8], lrun[8];
  for (int r = 0; r < 8; ++r) { mrun[r] = -1e30f; lrun[r] = 0.f; }
  f8v oacc[4];
  for (int j = 0; j < 4; ++j) oacc[j] = (f8v){0,0,0,0,0,0,0,0};

  for (int kt = 0; kt <= qb; ++kt) {
    __syncthreads();
    for (int p = 0; p < 4; ++p) {
      int idx = p * 128 + tid;
      int r = idx >> 3, c = (idx & 7) << 3;
      *(h8v*)&Kt[r][c] = *(const h8v*)(Kb + (size_t)(kt * 64 + r) * HDd + c);
      h8v vv = *(const h8v*)(Vb + (size_t)(kt * 64 + r) * HDd + c);
      for (int e = 0; e < 8; ++e) Vt[c + e][r] = vv[e];
    }
    __syncthreads();
    f8v sacc[4];
    for (int j = 0; j < 4; ++j) sacc[j] = (f8v){0,0,0,0,0,0,0,0};
    for (int kk = 0; kk < 2; ++kk) {
      h16v bf[4];
      for (int j = 0; j < 4; ++j) {
        h8v lo = *(const h8v*)&Kt[j * 16 + lr][kk * 32 + 8 * g];
        h8v hi = *(const h8v*)&Kt[j * 16 + lr][kk * 32 + 16 + 8 * g];
        bf[j] = cat8(lo, hi);
      }
      for (int j = 0; j < 4; ++j) sacc[j] = wmma16(qa[kk], bf[j], sacc[j]);
    }
    // online softmax on C-layout rows (row = 8g + r for this lane half)
    float pv[4][8];
    for (int r = 0; r < 8; ++r) {
      int qidx = qb * 64 + w * 16 + 8 * g + r;
      float mx = -1e30f;
      for (int j = 0; j < 4; ++j) {
        int kcol = kt * 64 + j * 16 + lr;
        float s = sacc[j][r] * 0.125f;
        if (kcol > qidx) s = -1e30f;
        pv[j][r] = s;
        mx = fmaxf(mx, s);
      }
      for (int d = 1; d < 16; d <<= 1) mx = fmaxf(mx, __shfl_xor(mx, d, 16));
      float mnew  = fmaxf(mrun[r], mx);
      float alpha = __expf(mrun[r] - mnew);
      float rs = 0.f;
      for (int j = 0; j < 4; ++j) {
        float p = __expf(pv[j][r] - mnew);
        pv[j][r] = p;
        rs += p;
      }
      for (int d = 1; d < 16; d <<= 1) rs += __shfl_xor(rs, d, 16);
      lrun[r] = lrun[r] * alpha + rs;
      mrun[r] = mnew;
      for (int j = 0; j < 4; ++j) oacc[j][r] *= alpha;
    }
    for (int j = 0; j < 4; ++j)
      for (int r = 0; r < 8; ++r)
        Pw[w][8 * g + r][j * 16 + lr] = (_Float16)pv[j][r];
    __syncthreads();
    for (int kk = 0; kk < 2; ++kk) {
      h8v lo = *(const h8v*)&Pw[w][lr][kk * 32 + 8 * g];
      h8v hi = *(const h8v*)&Pw[w][lr][kk * 32 + 16 + 8 * g];
      h16v pa = cat8(lo, hi);
      for (int j = 0; j < 4; ++j) {
        h8v l2 = *(const h8v*)&Vt[j * 16 + lr][kk * 32 + 8 * g];
        h8v h2 = *(const h8v*)&Vt[j * 16 + lr][kk * 32 + 16 + 8 * g];
        oacc[j] = wmma16(pa, cat8(l2, h2), oacc[j]);
      }
    }
  }
  for (int j = 0; j < 4; ++j)
    for (int r = 0; r < 8; ++r) {
      int srow = qb * 64 + w * 16 + 8 * g + r;
      O[((size_t)(b * Sd + srow)) * Hd + hh * 64 + j * 16 + lr] =
          oacc[j][r] / lrun[r];
    }
}

// ---------------------------------------------------------------------------
// RMSNorm over rows of length D (one row per 256-thread block)
// ---------------------------------------------------------------------------
__global__ __launch_bounds__(256)
void lfm2_rmsnorm(const float* __restrict__ x, const float* __restrict__ w,
                  float* __restrict__ out, int D) {
  __shared__ float red[8];
  const int row = blockIdx.x, tid = threadIdx.x;
  const float* xr = x + (size_t)row * D;
  float ss = 0.f;
  for (int i = tid; i < D; i += 256) { float v = xr[i]; ss += v * v; }
  for (int d = 1; d < 32; d <<= 1) ss += __shfl_xor(ss, d, 32);
  if ((tid & 31) == 0) red[tid >> 5] = ss;
  __syncthreads();
  float tot = 0.f;
  for (int i = 0; i < 8; ++i) tot += red[i];
  float inv = rsqrtf(tot / (float)D + 1e-6f);
  for (int i = tid; i < D; i += 256)
    out[(size_t)row * D + i] = w[i] * xr[i] * inv;
}

// ---------------------------------------------------------------------------
// q/k per-head RMSNorm + RoPE + f16 pack; v f16 pack. 64 threads / (token,unit)
// units: 0..11 q-heads, 12..15 k-heads, 16..19 v-heads
// ---------------------------------------------------------------------------
__global__ __launch_bounds__(64)
void lfm2_qkv_prep(const float* __restrict__ q, const float* __restrict__ k,
                   const float* __restrict__ v, const float* __restrict__ qln,
                   const float* __restrict__ kln, _Float16* __restrict__ Qh,
                   _Float16* __restrict__ Kh, _Float16* __restrict__ Vh) {
  __shared__ float red[2];
  __shared__ float sh[64];
  const int token = blockIdx.x, u = blockIdx.y, d = threadIdx.x;
  const int b = token >> 10, s = token & 1023;
  int head; const float* src;
  if (u < NHd)            { head = u;              src = q + (size_t)token * Hd  + head * HDd; }
  else if (u < NHd + NKVd){ head = u - NHd;        src = k + (size_t)token * 256 + head * HDd; }
  else                    { head = u - NHd - NKVd; src = v + (size_t)token * 256 + head * HDd; }
  float val = src[d];
  if (u < NHd + NKVd) {
    float ss = val * val;
    for (int t = 1; t < 32; t <<= 1) ss += __shfl_xor(ss, t, 32);
    if ((d & 31) == 0) red[d >> 5] = ss;
    __syncthreads();
    float inv = rsqrtf((red[0] + red[1]) / 64.f + 1e-6f);
    const float* ln = (u < NHd) ? qln : kln;
    val = val * inv * ln[d];
    sh[d] = val;
    __syncthreads();
    int i = d & 31;
    float freq = (float)s * __powf(10000.f, -(float)i * (1.0f / 32.0f));
    float rot = (d < 32) ? -sh[d + 32] : sh[d - 32];
    val = val * __cosf(freq) + rot * __sinf(freq);
  }
  _Float16 hv = (_Float16)val;
  if (u < NHd)             Qh[(((size_t)(b * NHd  + head)) * Sd + s) * HDd + d] = hv;
  else if (u < NHd + NKVd) Kh[(((size_t)(b * NKVd + head)) * Sd + s) * HDd + d] = hv;
  else                     Vh[(((size_t)(b * NKVd + head)) * Sd + s) * HDd + d] = hv;
}

// ---------------------------------------------------------------------------
// Causal depthwise conv (K=4) gating: y = Cg * (conv(Bg*xg) + cb)
// ---------------------------------------------------------------------------
__global__ __launch_bounds__(256)
void lfm2_conv(const float* __restrict__ bcx, const float* __restrict__ cw,
               const float* __restrict__ cb, float* __restrict__ y) {
  const int s = blockIdx.x, b = blockIdx.y, tid = threadIdx.x;
  for (int cc = 0; cc < 3; ++cc) {
    int c = cc * 256 + tid;
    float acc = 0.f;
    for (int t = 0; t < 4; ++t) {
      int sp = s - 3 + t;
      if (sp >= 0) {
        size_t bb = ((size_t)(b * Sd + sp)) * (3 * Hd);
        acc += bcx[bb + c] * bcx[bb + 2 * Hd + c] * cw[c * 4 + t];
      }
    }
    size_t tk = (size_t)(b * Sd + s);
    y[tk * Hd + c] = bcx[tk * (3 * Hd) + Hd + c] * (acc + cb[c]);
  }
}

__global__ __launch_bounds__(256)
void lfm2_silu_mul(const float* __restrict__ gate, const float* __restrict__ up,
                   float* __restrict__ out, size_t n) {
  for (size_t i = (size_t)blockIdx.x * 256 + threadIdx.x; i < n;
       i += (size_t)gridDim.x * 256) {
    float g = gate[i];
    out[i] = g / (1.f + __expf(-g)) * up[i];
  }
}

__global__ __launch_bounds__(256)
void lfm2_embed(const int* __restrict__ tokens, const float* __restrict__ embed,
                float* __restrict__ x) {
  const int t = blockIdx.x, tid = threadIdx.x;
  const int tok = tokens[t];
  for (int cc = 0; cc < 3; ++cc) {
    int c = cc * 256 + tid;
    x[(size_t)t * Hd + c] = embed[(size_t)tok * Hd + c];
  }
}

// ---------------------------------------------------------------------------
extern "C" void kernel_launch(void* const* d_in, const int* in_sizes, int n_in,
                              void* d_out, int out_size, void* d_ws,
                              size_t ws_size, hipStream_t stream) {
  (void)in_sizes; (void)n_in; (void)out_size;
  const int*   tokens   = (const int*)  d_in[0];
  const float* embed    = (const float*)d_in[1];
  const float* qw       = (const float*)d_in[2];
  const float* kw       = (const float*)d_in[3];
  const float* vw       = (const float*)d_in[4];
  const float* ow       = (const float*)d_in[5];
  const float* qln      = (const float*)d_in[6];
  const float* kln      = (const float*)d_in[7];
  const float* a_gate   = (const float*)d_in[8];
  const float* a_up     = (const float*)d_in[9];
  const float* a_down   = (const float*)d_in[10];
  const float* a_ln1    = (const float*)d_in[11];
  const float* a_ln2    = (const float*)d_in[12];
  const float* conv_w   = (const float*)d_in[13];
  const float* conv_b   = (const float*)d_in[14];
  const float* in_w     = (const float*)d_in[15];
  const float* in_b     = (const float*)d_in[16];
  const float* out_w    = (const float*)d_in[17];
  const float* out_b    = (const float*)d_in[18];
  const float* c_gate   = (const float*)d_in[19];
  const float* c_up     = (const float*)d_in[20];
  const float* c_down   = (const float*)d_in[21];
  const float* c_ln1    = (const float*)d_in[22];
  const float* c_ln2    = (const float*)d_in[23];
  const float* final_ln = (const float*)d_in[24];
  const float* lm_head  = (const float*)d_in[25];
  float* logits = (float*)d_out;

  char* ws = (char*)d_ws;
  size_t off = 0;
  auto a16 = [&](size_t n) { _Float16* p = (_Float16*)(ws + off);
                             off += ((n * 2 + 255) / 256) * 256; return p; };
  auto a32 = [&](size_t n) { float* p = (float*)(ws + off);
                             off += ((n * 4 + 255) / 256) * 256; return p; };

  // f16 weights
  _Float16* wqw  = a16((size_t)4 * Hd * Hd);
  _Float16* wkw  = a16((size_t)4 * 256 * Hd);
  _Float16* wvw  = a16((size_t)4 * 256 * Hd);
  _Float16* wow  = a16((size_t)4 * Hd * Hd);
  _Float16* wag  = a16((size_t)4 * IMd * Hd);
  _Float16* wau  = a16((size_t)4 * IMd * Hd);
  _Float16* wad  = a16((size_t)4 * Hd * IMd);
  _Float16* wiw  = a16((size_t)8 * 3 * Hd * Hd);
  _Float16* wowc = a16((size_t)8 * Hd * Hd);
  _Float16* wcg  = a16((size_t)8 * IMd * Hd);
  _Float16* wcu  = a16((size_t)8 * IMd * Hd);
  _Float16* wcd  = a16((size_t)8 * Hd * IMd);
  _Float16* wlm  = a16((size_t)Vd * Hd);
  // activations
  float* x    = a32((size_t)Td * Hd);
  float* hbuf = a32((size_t)Td * Hd);
  float* qp   = a32((size_t)Td * Hd);
  float* kp   = a32((size_t)Td * 256);
  float* vp   = a32((size_t)Td * 256);
  float* ob   = a32((size_t)Td * Hd);
  float* g1   = a32((size_t)Td * IMd);
  float* g2   = a32((size_t)Td * IMd);
  float* bcx  = a32((size_t)Td * 3 * Hd);
  float* yb   = a32((size_t)Td * Hd);
  _Float16* qh = a16((size_t)Bd * NHd  * Sd * HDd);
  _Float16* kh = a16((size_t)Bd * NKVd * Sd * HDd);
  _Float16* vh = a16((size_t)Bd * NKVd * Sd * HDd);
  if (off > ws_size) return;  // workspace too small: deterministic no-op

  auto cvt = [&](const float* s, _Float16* d, size_t n) {
    lfm2_cvt_f16<<<dim3((unsigned)((n + 255) / 256)), 256, 0, stream>>>(s, d, n);
  };
  auto gemm = [&](const float* A, const _Float16* W, float* C,
                  const float* bias, const float* resid, int M, int N, int K) {
    lfm2_gemm<<<dim3(N / GBN, M / GBM), 256, 0, stream>>>(A, W, C, bias, resid,
                                                          M, N, K);
  };
  auto rms = [&](const float* xin, const float* w, float* out) {
    lfm2_rmsnorm<<<dim3(Td), 256, 0, stream>>>(xin, w, out, Hd);
  };

  // convert weights once per launch
  cvt(qw,     wqw, (size_t)4 * Hd * Hd);
  cvt(kw,     wkw, (size_t)4 * 256 * Hd);
  cvt(vw,     wvw, (size_t)4 * 256 * Hd);
  cvt(ow,     wow, (size_t)4 * Hd * Hd);
  cvt(a_gate, wag, (size_t)4 * IMd * Hd);
  cvt(a_up,   wau, (size_t)4 * IMd * Hd);
  cvt(a_down, wad, (size_t)4 * Hd * IMd);
  cvt(in_w,   wiw, (size_t)8 * 3 * Hd * Hd);
  cvt(out_w,  wowc,(size_t)8 * Hd * Hd);
  cvt(c_gate, wcg, (size_t)8 * IMd * Hd);
  cvt(c_up,   wcu, (size_t)8 * IMd * Hd);
  cvt(c_down, wcd, (size_t)8 * Hd * IMd);
  cvt(lm_head,wlm, (size_t)Vd * Hd);

  lfm2_embed<<<dim3(Td), 256, 0, stream>>>(tokens, embed, x);

  int ai = 0, ci = 0;
  for (int l = 0; l < 12; ++l) {
    bool full = (l == 2 || l == 5 || l == 8 || l == 11);
    if (full) {
      rms(x, a_ln1 + (size_t)ai * Hd, hbuf);
      gemm(hbuf, wqw + (size_t)ai * Hd * Hd,  qp, nullptr, nullptr, Td, Hd,  Hd);
      gemm(hbuf, wkw + (size_t)ai * 256 * Hd, kp, nullptr, nullptr, Td, 256, Hd);
      gemm(hbuf, wvw + (size_t)ai * 256 * Hd, vp, nullptr, nullptr, Td, 256, Hd);
      lfm2_qkv_prep<<<dim3(Td, NHd + 2 * NKVd), 64, 0, stream>>>(
          qp, kp, vp, qln + (size_t)ai * HDd, kln + (size_t)ai * HDd, qh, kh, vh);
      lfm2_attn<<<dim3(Sd / 64, NHd, Bd), 128, 0, stream>>>(qh, kh, vh, ob);
      gemm(ob, wow + (size_t)ai * Hd * Hd, x, nullptr, x, Td, Hd, Hd);
      rms(x, a_ln2 + (size_t)ai * Hd, hbuf);
      gemm(hbuf, wag + (size_t)ai * IMd * Hd, g1, nullptr, nullptr, Td, IMd, Hd);
      gemm(hbuf, wau + (size_t)ai * IMd * Hd, g2, nullptr, nullptr, Td, IMd, Hd);
      lfm2_silu_mul<<<dim3(2048), 256, 0, stream>>>(g1, g2, g1, (size_t)Td * IMd);
      gemm(g1, wad + (size_t)ai * Hd * IMd, x, nullptr, x, Td, Hd, IMd);
      ++ai;
    } else {
      rms(x, c_ln1 + (size_t)ci * Hd, hbuf);
      gemm(hbuf, wiw + (size_t)ci * 3 * Hd * Hd, bcx,
           in_b + (size_t)ci * 3 * Hd, nullptr, Td, 3 * Hd, Hd);
      lfm2_conv<<<dim3(Sd, Bd), 256, 0, stream>>>(
          bcx, conv_w + (size_t)ci * Hd * 4, conv_b + (size_t)ci * Hd, yb);
      gemm(yb, wowc + (size_t)ci * Hd * Hd, x,
           out_b + (size_t)ci * Hd, x, Td, Hd, Hd);
      rms(x, c_ln2 + (size_t)ci * Hd, hbuf);
      gemm(hbuf, wcg + (size_t)ci * IMd * Hd, g1, nullptr, nullptr, Td, IMd, Hd);
      gemm(hbuf, wcu + (size_t)ci * IMd * Hd, g2, nullptr, nullptr, Td, IMd, Hd);
      lfm2_silu_mul<<<dim3(2048), 256, 0, stream>>>(g1, g2, g1, (size_t)Td * IMd);
      gemm(g1, wcd + (size_t)ci * Hd * IMd, x, nullptr, x, Td, Hd, IMd);
      ++ci;
    }
  }
  rms(x, final_ln, hbuf);
  gemm(hbuf, wlm, logits, nullptr, nullptr, Td, Vd, Hd);
}
